// TextEncoderGenerator_4475355923026
// MI455X (gfx1250) — matrix operand order repaired and checked by reading, not compiled, not run on hardware
//
#include <hip/hip_runtime.h>
#include <hip/hip_bf16.h>

// ---------------- problem constants ----------------
constexpr int BATCH = 256;
constexpr int SEQ   = 128;
constexpr int IN    = 300;
constexpr int INP   = 320;   // IN padded to multiple of 32 for WMMA K
constexpr int H     = 512;
constexpr int G3    = 3 * H; // 1536
constexpr int COND  = 128;

typedef __attribute__((ext_vector_type(8)))  __bf16 v8bf;
typedef __attribute__((ext_vector_type(16))) __bf16 v16bf;
typedef __attribute__((ext_vector_type(8)))  float  v8f;

// ---------------- helpers ----------------
__device__ inline v8f wmma_bf16(v16bf a, v16bf b, v8f c) {
  // (neg_a, A, neg_b, B, c_mod, C, reuse_a, reuse_b)
  return __builtin_amdgcn_wmma_f32_16x16x32_bf16(false, a, false, b, (short)0, c,
                                                 false, false);
}

// A-fragment (16x32 bf16): lane m = lane&15, half = lane>>4.
// slots[0..7]  = K k0+8*half .. +7 ; slots[8..15] = K k0+8*half+16 .. +23
__device__ inline v16bf load_a_frag(const __bf16* __restrict__ rowBase, int k0, int half) {
  v8bf lo = *(const v8bf*)(rowBase + k0 + 8 * half);
  v8bf hi = *(const v8bf*)(rowBase + k0 + 8 * half + 16);
  v16bf a;
#pragma unroll
  for (int j = 0; j < 8; ++j) { a[j] = lo[j]; a[j + 8] = hi[j]; }
  return a;
}

__device__ inline float sigmoidf_(float x) { return 1.0f / (1.0f + __expf(-x)); }

// ---------------- setup kernels ----------------
// dst[r*dstK + k] = (k < srcK) ? bf16(src[r*srcK + k]) : 0
__global__ void cvt_pad_bf16(const float* __restrict__ src, __bf16* __restrict__ dst,
                             int rows, int srcK, int dstK) {
  int idx = blockIdx.x * blockDim.x + threadIdx.x;
  int total = rows * dstK;
  if (idx >= total) return;
  int r = idx / dstK, k = idx - r * dstK;
  float v = (k < srcK) ? src[r * srcK + k] : 0.0f;
  dst[idx] = (__bf16)v;
}

__global__ void zero_u32(unsigned* __restrict__ p, int n) {
  int idx = blockIdx.x * blockDim.x + threadIdx.x;
  if (idx < n) p[idx] = 0u;
}

// ---------------- fused GRU step ----------------
// grid: (H/16=32, BATCH/16=16, 2 dirs), block: 128 (4 waves)
// wave0: r preact (x:K=320 fused h:K=512), wave1: z preact, wave2: i_n, wave3: h_n
__global__ __launch_bounds__(128) void gru_step(
    int step,
    const __bf16* __restrict__ Xbf,
    const __bf16* __restrict__ WihF, const __bf16* __restrict__ WhhF,
    const __bf16* __restrict__ WihB, const __bf16* __restrict__ WhhB,
    const float* __restrict__ bihF, const float* __restrict__ bhhF,
    const float* __restrict__ bihB, const float* __restrict__ bhhB,
    const int* __restrict__ lens,
    const __bf16* __restrict__ HbfIn, const float* __restrict__ HfIn,
    __bf16* __restrict__ HbfOut, float* __restrict__ HfOut,
    float* __restrict__ accF, float* __restrict__ accB) {
  const int dir   = blockIdx.z;
  const int nTile = blockIdx.x;
  const int mTile = blockIdx.y;
  const int tid   = threadIdx.x;
  const int w     = tid >> 5;
  const int lane  = tid & 31;
  const int half  = lane >> 4;
  const int l15   = lane & 15;

  const __bf16* Wih = dir ? WihB : WihF;
  const __bf16* Whh = dir ? WhhB : WhhF;
  const float*  bih = dir ? bihB : bihF;
  const float*  bhh = dir ? bhhB : bhhF;
  float*        acc = dir ? accB : accF;
  const __bf16* hbfIn = HbfIn + (size_t)dir * BATCH * H;
  const float*  hfIn  = HfIn  + (size_t)dir * BATCH * H;
  __bf16* hbfOut = HbfOut + (size_t)dir * BATCH * H;
  float*  hfOut  = HfOut  + (size_t)dir * BATCH * H;

  const int xt   = dir ? (SEQ - 1 - step) : step;
  const int mRow = mTile * 16 + l15;                 // batch row for A
  const int gate = (w < 2) ? w : 2;
  const int wRow = gate * H + nTile * 16 + l15;      // weight row = B column

  __shared__ float lds[4][16][16];

  v8f c = {};

  if (w != 3) {  // x @ Wih^T contribution
    const __bf16* aBase = Xbf + ((size_t)mRow * SEQ + xt) * INP;
    const __bf16* bBase = Wih + (size_t)wRow * INP;
#pragma unroll
    for (int kc = 0; kc < INP / 32; ++kc) {
      const int k0 = kc * 32;
      v16bf a = load_a_frag(aBase, k0, half);
      v16bf b = load_a_frag(bBase, k0, half);
      c = wmma_bf16(a, b, c);
    }
  }
  if (w != 2) {  // h @ Whh^T contribution
    const __bf16* aBase = hbfIn + (size_t)mRow * H;
    const __bf16* bBase = Whh + (size_t)wRow * H;
#pragma unroll
    for (int kc = 0; kc < H / 32; ++kc) {
      const int k0 = kc * 32;
      v16bf a = load_a_frag(aBase, k0, half);
      v16bf b = load_a_frag(bBase, k0, half);
      c = wmma_bf16(a, b, c);
    }
  }

  // C/D layout: lane col = l15, rows 8*half + r
#pragma unroll
  for (int r = 0; r < 8; ++r) lds[w][8 * half + r][l15] = c[r];
  __syncthreads();

  // gate nonlinearity + state update: 256 elems / 128 threads
#pragma unroll
  for (int e = tid; e < 256; e += 128) {
    const int m = e >> 4, n = e & 15;
    const int row = mTile * 16 + m;
    const int col = nTile * 16 + n;
    const float pr  = lds[0][m][n] + bih[col]         + bhh[col];
    const float pz  = lds[1][m][n] + bih[H + col]     + bhh[H + col];
    const float xin = lds[2][m][n] + bih[2 * H + col];
    const float hn  = lds[3][m][n] + bhh[2 * H + col];
    const float r = sigmoidf_(pr);
    const float z = sigmoidf_(pz);
    const float nn = tanhf(xin + r * hn);
    const float hp = hfIn[(size_t)row * H + col];
    const float hnew = (1.0f - z) * nn + z * hp;
    hfOut[(size_t)row * H + col]  = hnew;
    hbfOut[(size_t)row * H + col] = (__bf16)hnew;
    if (step < lens[row]) acc[(size_t)row * H + col] += hnew;
  }
}

// ---------------- finalize ----------------
__global__ void finalize_avg(const float* __restrict__ aF, const float* __restrict__ aB,
                             __bf16* __restrict__ avgbf, int n) {
  int idx = blockIdx.x * blockDim.x + threadIdx.x;
  if (idx < n) avgbf[idx] = (__bf16)((aF[idx] + aB[idx]) * (0.5f / (float)SEQ));
}

// grid: (COND/16=8, BATCH/16=16), 1 wave per block. out = mu*(1+noise), mu = lrelu(avg@Wmu^T + b)
__global__ __launch_bounds__(32) void proj_out(
    const __bf16* __restrict__ avgbf, const __bf16* __restrict__ Wmu,
    const float* __restrict__ bmu, const float* __restrict__ noise,
    float* __restrict__ out) {
  const int nTile = blockIdx.x;
  const int mTile = blockIdx.y;
  const int lane = threadIdx.x & 31;
  const int half = lane >> 4;
  const int l15  = lane & 15;

  v8f c = {};
  const __bf16* aBase = avgbf + (size_t)(mTile * 16 + l15) * H;
  const __bf16* bBase = Wmu + (size_t)(nTile * 16 + l15) * H;
#pragma unroll
  for (int kc = 0; kc < H / 32; ++kc) {
    const int k0 = kc * 32;
    v16bf a = load_a_frag(aBase, k0, half);
    v16bf b = load_a_frag(bBase, k0, half);
    c = wmma_bf16(a, b, c);
  }
#pragma unroll
  for (int r = 0; r < 8; ++r) {
    const int row = mTile * 16 + 8 * half + r;
    const int col = nTile * 16 + l15;
    const float v = c[r] + bmu[col];
    const float mu = (v > 0.0f) ? v : 0.2f * v;
    out[(size_t)row * COND + col] = mu * (1.0f + noise[(size_t)row * COND + col]);
  }
}

// ---------------- host ----------------
extern "C" void kernel_launch(void* const* d_in, const int* in_sizes, int n_in,
                              void* d_out, int out_size, void* d_ws, size_t ws_size,
                              hipStream_t stream) {
  (void)in_sizes; (void)n_in; (void)out_size; (void)ws_size;
  const float* text  = (const float*)d_in[0];
  const int*   lens  = (const int*)d_in[1];
  const float* Wih_f = (const float*)d_in[2];
  const float* Whh_f = (const float*)d_in[3];
  const float* bih_f = (const float*)d_in[4];
  const float* bhh_f = (const float*)d_in[5];
  const float* Wih_b = (const float*)d_in[6];
  const float* Whh_b = (const float*)d_in[7];
  const float* bih_b = (const float*)d_in[8];
  const float* bhh_b = (const float*)d_in[9];
  const float* W_mu  = (const float*)d_in[10];
  const float* b_mu  = (const float*)d_in[11];
  const float* noise = (const float*)d_in[12];
  float* out = (float*)d_out;

  char* ws = (char*)d_ws;
  size_t off = 0;
  auto alloc = [&](size_t bytes) {
    size_t o = off;
    off = (off + bytes + 255) & ~(size_t)255;
    return o;
  };
  const size_t oXbf  = alloc((size_t)BATCH * SEQ * INP * 2);
  const size_t oWihF = alloc((size_t)G3 * INP * 2);
  const size_t oWihB = alloc((size_t)G3 * INP * 2);
  const size_t oWhhF = alloc((size_t)G3 * H * 2);
  const size_t oWhhB = alloc((size_t)G3 * H * 2);
  const size_t oWmu  = alloc((size_t)COND * H * 2);
  const size_t oHbf  = alloc((size_t)2 * 2 * BATCH * H * 2);  // [parity][dir]
  const size_t oHf   = alloc((size_t)2 * 2 * BATCH * H * 4);
  const size_t oAccF = alloc((size_t)BATCH * H * 4);
  const size_t oAccB = alloc((size_t)BATCH * H * 4);
  const size_t oAvg  = alloc((size_t)BATCH * H * 2);

  __bf16* Xbf   = (__bf16*)(ws + oXbf);
  __bf16* WihFb = (__bf16*)(ws + oWihF);
  __bf16* WihBb = (__bf16*)(ws + oWihB);
  __bf16* WhhFb = (__bf16*)(ws + oWhhF);
  __bf16* WhhBb = (__bf16*)(ws + oWhhB);
  __bf16* WmuB  = (__bf16*)(ws + oWmu);
  __bf16* Hbf   = (__bf16*)(ws + oHbf);
  float*  Hf    = (float*)(ws + oHf);
  float*  accF  = (float*)(ws + oAccF);
  float*  accB  = (float*)(ws + oAccB);
  __bf16* avgbf = (__bf16*)(ws + oAvg);

  const int T = 256;
  auto blk = [&](int n) { return (n + T - 1) / T; };

  // --- setup: bf16 conversions (pad K 300 -> 320) ---
  cvt_pad_bf16<<<blk(BATCH * SEQ * INP), T, 0, stream>>>(text, Xbf, BATCH * SEQ, IN, INP);
  cvt_pad_bf16<<<blk(G3 * INP), T, 0, stream>>>(Wih_f, WihFb, G3, IN, INP);
  cvt_pad_bf16<<<blk(G3 * INP), T, 0, stream>>>(Wih_b, WihBb, G3, IN, INP);
  cvt_pad_bf16<<<blk(G3 * H), T, 0, stream>>>(Whh_f, WhhFb, G3, H, H);
  cvt_pad_bf16<<<blk(G3 * H), T, 0, stream>>>(Whh_b, WhhBb, G3, H, H);
  cvt_pad_bf16<<<blk(COND * H), T, 0, stream>>>(W_mu, WmuB, COND, H, H);

  // --- zero state + accumulators (ws is poisoned; must re-zero every call) ---
  zero_u32<<<blk(2 * 2 * BATCH * H / 2), T, 0, stream>>>((unsigned*)Hbf, 2 * 2 * BATCH * H / 2);
  zero_u32<<<blk(2 * 2 * BATCH * H), T, 0, stream>>>((unsigned*)Hf, 2 * 2 * BATCH * H);
  zero_u32<<<blk(BATCH * H), T, 0, stream>>>((unsigned*)accF, BATCH * H);
  zero_u32<<<blk(BATCH * H), T, 0, stream>>>((unsigned*)accB, BATCH * H);

  // --- 128 recurrent steps, fused fwd+bwd, ping-pong h ---
  const size_t pstride = (size_t)2 * BATCH * H;  // [dir] pair per parity
  for (int s = 0; s < SEQ; ++s) {
    const int pin = s & 1, pout = pin ^ 1;
    gru_step<<<dim3(H / 16, BATCH / 16, 2), 128, 0, stream>>>(
        s, Xbf, WihFb, WhhFb, WihBb, WhhBb, bih_f, bhh_f, bih_b, bhh_b, lens,
        Hbf + pin * pstride, Hf + pin * pstride,
        Hbf + pout * pstride, Hf + pout * pstride,
        accF, accB);
  }

  // --- mean over time, projection, leaky-relu, noise ---
  finalize_avg<<<blk(BATCH * H), T, 0, stream>>>(accF, accB, avgbf, BATCH * H);
  proj_out<<<dim3(COND / 16, BATCH / 16), 32, 0, stream>>>(avgbf, WmuB, b_mu, noise, out);
}
